// CrossDimEncoder_77618648973583
// MI455X (gfx1250) — matrix operand order repaired and checked by reading, not compiled, not run on hardware
//
#include <hip/hip_runtime.h>

// ---------------------------------------------------------------------------
// CDNA5 (gfx1250) fused GNN encoder. wave32, WMMA f32_16x16x32_f16.
// Tile = one graph node x 16 batch items -> gather indices are wave-uniform
// (scalarized via readfirstlane so EXEC provably stays all-ones around WMMA).
// ---------------------------------------------------------------------------

typedef __attribute__((ext_vector_type(16))) _Float16 v16h;
typedef __attribute__((ext_vector_type(8)))  float    v8f;

static constexpr int Bsz = 2048, Tt = 19, Nn = 54, Hd = 128;

// ---- workspace layout (units: _Float16 elements) ----
static constexpr size_t OFF_H16  = 0;
static constexpr size_t OFF_V16  = OFF_H16 + (size_t)Bsz*Tt*Hd;       // h activations f16
static constexpr size_t OFF_V1   = OFF_V16 + (size_t)Bsz*Nn*Hd;       // v activations f16
static constexpr size_t OFF_CMB  = OFF_V1  + (size_t)Bsz*Nn*Hd;       // post-inflate v
static constexpr size_t OFF_WBLK = OFF_CMB + (size_t)Bsz*448;         // pooled features
static constexpr size_t WBLK_SZ  = 245760;                            // f16 weights / block
static constexpr size_t WO_INF=0, WO_MSG=49152, WO_UPD=65536, WO_DEF=98304,
                        WO_VF1=114688, WO_VF2=147456, WO_HF1=180224, WO_HF2=212992;
static constexpr size_t OFF_OUT1 = OFF_WBLK + 4*WBLK_SZ;
static constexpr size_t OFF_OUT2 = OFF_OUT1 + (size_t)448*256;

// ---- d_in flattening (top-level insertion order; params dict sorted-key, jax-style) ----
enum { DEF_W=0, DEF_B, HF1_W, HF1_B, HF2_W, HF2_B, HN_B, HN_G, INF_W, INF_B,
       MSG_W, MSG_B, UPD_W, UPD_B, VF1_W, VF1_B, VF2_W, VF2_B, VN_B, VN_G };
static constexpr int IDX_HEXF=0, IDX_VERTF=1, IDX_EDGEF=2, IDX_PLAYERF=3;
static constexpr int IDX_BLK0=4;                                   // 4 blocks x 20 leaves
static constexpr int IDX_EMLP_W=84, IDX_EMLP_B=85, IDX_EPROJ_W=86, IDX_EPROJ_B=87;
static constexpr int IDX_HPROJ_W=88, IDX_HPROJ_B=89, IDX_OUT1_W=90, IDX_OUT1_B=91;
static constexpr int IDX_OUT2_W=92, IDX_OUT2_B=93, IDX_P1_W=94, IDX_P1_B=95;
static constexpr int IDX_P2_W=96, IDX_P2_B=97, IDX_VPROJ_W=98, IDX_VPROJ_B=99;
static constexpr int IDX_V2H=100, IDX_H2V=101, IDX_ADJ=102;

// ---------------------------------------------------------------------------
// fragment helpers (documented CDNA5 16-bit WMMA layouts)
// ---------------------------------------------------------------------------
__device__ __forceinline__ v8f wmma(v16h a, v16h b, v8f c) {
  return __builtin_amdgcn_wmma_f32_16x16x32_f16(false, a, false, b, (short)0, c, false, false);
}
// A 16x32 f16: lanes<16 rows 0..15 hold K {kb..kb+7, 16+kb..16+kb+7}, kb=0; lanes>=16 kb=8.
__device__ __forceinline__ v16h load_af(const _Float16* row) {
  int lane = threadIdx.x & 31;
  int kb = (lane & 16) ? 8 : 0;
  union { int4 q[2]; v16h v; } u;
  u.q[0] = *reinterpret_cast<const int4*>(row + kb);
  u.q[1] = *reinterpret_cast<const int4*>(row + 16 + kb);
  return u.v;
}
// B 32x16 f16 from transposed weights Wt[n][k]: lane holds column n, K {kb..kb+7, 16+kb..16+kb+7}, kb=0 / 8.
__device__ __forceinline__ v16h load_bf(const _Float16* Wt, int K, int nt, int kofs) {
  int lane = threadIdx.x & 31;
  int col = lane & 15;
  int ko  = (lane & 16) ? 8 : 0;
  const _Float16* p = Wt + (size_t)(nt*16 + col) * K + kofs + ko;
  union { int4 q[2]; v16h v; } u;
  u.q[0] = *reinterpret_cast<const int4*>(p);
  u.q[1] = *reinterpret_cast<const int4*>(p + 16);
  return u.v;
}
__device__ __forceinline__ float hred16(float x) {     // reduce within 16-lane halves
  x += __shfl_xor(x, 1, 32); x += __shfl_xor(x, 2, 32);
  x += __shfl_xor(x, 4, 32); x += __shfl_xor(x, 8, 32);
  return x;
}
__device__ __forceinline__ float gelu_f(float x) {
  return 0.5f * x * (1.0f + erff(x * 0.70710678118654752f));
}
__device__ __forceinline__ void lds_fence() {
  asm volatile("s_wait_dscnt 0" ::: "memory");
}
__device__ __forceinline__ void prefetch_w(const _Float16* Wt, int K) {
  // warm a few weight lines (global_prefetch_b8); one line per output n-tile row
  int col = threadIdx.x & 15;
  #pragma unroll
  for (int nt = 0; nt < 8; ++nt)
    __builtin_prefetch(Wt + (size_t)(nt*16 + col) * K, 0, 1);
}

// ---------------------------------------------------------------------------
// weight conversion: f32 (K,N) -> f16 transposed Wt (N,K)
// ---------------------------------------------------------------------------
static constexpr int NCONV = 34;
struct ConvDesc { const float* src; unsigned dst_off; int K; int N; };
struct ConvArgs { ConvDesc d[NCONV]; };

__global__ __launch_bounds__(256) void k_convert(ConvArgs a, _Float16* ws) {
  ConvDesc cd = a.d[blockIdx.x];
  _Float16* dst = ws + cd.dst_off;
  int total = cd.K * cd.N;
  for (int i = threadIdx.x; i < total; i += 256) {
    int n = i / cd.K, k = i - n * cd.K;
    dst[i] = (_Float16)cd.src[(size_t)k * cd.N + n];
  }
}

// ---------------------------------------------------------------------------
// input projections (K=16, trivial FLOPs -> VALU)
// ---------------------------------------------------------------------------
__global__ __launch_bounds__(256) void k_proj(
    const float* __restrict__ hexf, const float* __restrict__ vertf,
    const float* __restrict__ hW, const float* __restrict__ hB,
    const float* __restrict__ vW, const float* __restrict__ vB,
    _Float16* __restrict__ h16, _Float16* __restrict__ v16) {
  size_t idx = (size_t)blockIdx.x * 256 + threadIdx.x;
  const size_t HTOT = (size_t)Bsz*Tt*Hd, VTOT = (size_t)Bsz*Nn*Hd;
  if (idx < HTOT) {
    size_t r = idx >> 7; int c = (int)(idx & 127);
    const float* f = hexf + r*16;
    float a = hB[c];
    #pragma unroll
    for (int k=0;k<16;++k) a += f[k]*hW[k*128 + c];
    h16[idx] = (_Float16)a;
  } else if (idx < HTOT + VTOT) {
    size_t i2 = idx - HTOT, r = i2 >> 7; int c = (int)(i2 & 127);
    const float* f = vertf + r*16;
    float a = vB[c];
    #pragma unroll
    for (int k=0;k<16;++k) a += f[k]*vW[k*128 + c];
    v16[i2] = (_Float16)a;
  }
}

// ---------------------------------------------------------------------------
// inflate: v1 = v + concat(masked gather h x3) @ inf_W + b    (WMMA)
// ---------------------------------------------------------------------------
__global__ __launch_bounds__(256) void k_inflate(
    const _Float16* __restrict__ h16, const _Float16* __restrict__ v16,
    _Float16* __restrict__ v1, const _Float16* __restrict__ infWt,
    const float* __restrict__ inf_b, const int* __restrict__ v2h) {
  int wave = threadIdx.x >> 5, lane = threadIdx.x & 31;
  int tile = blockIdx.x * 8 + wave;
  int bb = tile / Nn, node = tile - bb * Nn;
  int brow = lane & 15, m0 = (lane & 16) ? 8 : 0, col = lane & 15;
  prefetch_w(infWt, 384);
  v8f acc[8] = {};
  #pragma unroll
  for (int j = 0; j < 3; ++j) {
    int hx = __builtin_amdgcn_readfirstlane(v2h[node*3 + j]);  // wave-uniform -> SGPR
    if (hx >= 0) {
      const _Float16* rowp = h16 + ((size_t)(bb*16 + brow)*Tt + hx)*Hd;
      #pragma unroll
      for (int kt = 0; kt < 4; ++kt) {
        v16h A = load_af(rowp + kt*32);
        v16h Bf[8];
        #pragma unroll
        for (int nt = 0; nt < 8; ++nt) Bf[nt] = load_bf(infWt, 384, nt, j*128 + kt*32);
        #pragma unroll
        for (int nt = 0; nt < 8; ++nt) acc[nt] = wmma(A, Bf[nt], acc[nt]);
      }
    }
  }
  #pragma unroll
  for (int nt = 0; nt < 8; ++nt) {
    int c = nt*16 + col;
    float bias = inf_b[c];
    #pragma unroll
    for (int r = 0; r < 8; ++r) {
      size_t ridx = ((size_t)(bb*16 + m0 + r)*Nn + node)*Hd + c;
      v1[ridx] = (_Float16)(acc[nt][r] + bias + (float)v16[ridx]);
    }
  }
}

// ---------------------------------------------------------------------------
// vertex: msg((sum masked nb)@W + 3b)/cnt -> upd(concat) -> LN -> FFN (fused)
// ---------------------------------------------------------------------------
__global__ __launch_bounds__(256) void k_vertex_upd(
    const _Float16* __restrict__ v1, _Float16* __restrict__ vout,
    const _Float16* __restrict__ msgWt, const float* __restrict__ msg_b,
    const _Float16* __restrict__ updWt, const float* __restrict__ upd_b,
    const float* __restrict__ vn_g, const float* __restrict__ vn_b,
    const _Float16* __restrict__ vf1Wt, const float* __restrict__ vf1_b,
    const _Float16* __restrict__ vf2Wt, const float* __restrict__ vf2_b,
    const int* __restrict__ adj) {
  __shared__ _Float16 vtile[8][16*136];   // per-wave post-LN tile (A source for FFN)
  __shared__ _Float16 bnc[8][16*40];      // per-wave C->A bounce tile
  int wave = threadIdx.x >> 5, lane = threadIdx.x & 31;
  int tile = blockIdx.x * 8 + wave;
  int bb = tile / Nn, node = tile - bb * Nn;
  int brow = lane & 15, m0 = (lane & 16) ? 8 : 0, col = lane & 15;
  _Float16* vt = &vtile[wave][0];
  _Float16* bc = &bnc[wave][0];
  prefetch_w(msgWt, 128);
  prefetch_w(updWt, 256);

  // ---- msg: agg = ((sum masked neighbors)@W + 3b) / cnt ----
  int nb0 = __builtin_amdgcn_readfirstlane(adj[node*3+0]);
  int nb1 = __builtin_amdgcn_readfirstlane(adj[node*3+1]);
  int nb2 = __builtin_amdgcn_readfirstlane(adj[node*3+2]);
  int cnt = (nb0>=0) + (nb1>=0) + (nb2>=0);
  float rc = 1.0f / (float)(cnt > 1 ? cnt : 1);
  v8f accA[8] = {};
  #pragma unroll
  for (int kt = 0; kt < 4; ++kt) {
    v16h A = {};
    if (nb0>=0) A = A + load_af(v1 + ((size_t)(bb*16+brow)*Nn + nb0)*Hd + kt*32);
    if (nb1>=0) A = A + load_af(v1 + ((size_t)(bb*16+brow)*Nn + nb1)*Hd + kt*32);
    if (nb2>=0) A = A + load_af(v1 + ((size_t)(bb*16+brow)*Nn + nb2)*Hd + kt*32);
    v16h Bf[8];
    #pragma unroll
    for (int nt = 0; nt < 8; ++nt) Bf[nt] = load_bf(msgWt, 128, nt, kt*32);
    #pragma unroll
    for (int nt = 0; nt < 8; ++nt) accA[nt] = wmma(A, Bf[nt], accA[nt]);
  }
  // ---- upd: [v | agg] @ upd_W ----
  const _Float16* vrow = v1 + ((size_t)(bb*16+brow)*Nn + node)*Hd;
  v8f accU[8] = {};
  #pragma unroll
  for (int kt = 0; kt < 4; ++kt) {
    v16h A = load_af(vrow + kt*32);
    v16h Bf[8];
    #pragma unroll
    for (int nt = 0; nt < 8; ++nt) Bf[nt] = load_bf(updWt, 256, nt, kt*32);
    #pragma unroll
    for (int nt = 0; nt < 8; ++nt) accU[nt] = wmma(A, Bf[nt], accU[nt]);
  }
  #pragma unroll
  for (int p = 0; p < 4; ++p) {                    // agg half of K, via LDS bounce
    #pragma unroll
    for (int e = 0; e < 2; ++e) {
      int nt = 2*p + e;
      float mb = 3.0f * msg_b[nt*16 + col];
      #pragma unroll
      for (int r = 0; r < 8; ++r)
        bc[(m0+r)*40 + e*16 + col] = (_Float16)((accA[nt][r] + mb) * rc);
    }
    lds_fence();
    v16h A2 = load_af(bc + (size_t)brow*40);
    v16h Bf[8];
    #pragma unroll
    for (int nt = 0; nt < 8; ++nt) Bf[nt] = load_bf(updWt, 256, nt, 128 + p*32);
    #pragma unroll
    for (int nt = 0; nt < 8; ++nt) accU[nt] = wmma(A2, Bf[nt], accU[nt]);
  }
  // ---- residual + bias, then LayerNorm (row = batch item) ----
  #pragma unroll
  for (int nt = 0; nt < 8; ++nt) {
    float ub = upd_b[nt*16 + col];
    #pragma unroll
    for (int r = 0; r < 8; ++r)
      accU[nt][r] += ub + (float)v1[((size_t)(bb*16+m0+r)*Nn + node)*Hd + nt*16 + col];
  }
  float s1[8] = {}, s2[8] = {};
  #pragma unroll
  for (int nt = 0; nt < 8; ++nt)
    #pragma unroll
    for (int r = 0; r < 8; ++r) { float v = accU[nt][r]; s1[r] += v; s2[r] += v*v; }
  float mean[8], rstd[8];
  #pragma unroll
  for (int r = 0; r < 8; ++r) {
    float su = hred16(s1[r]), sq = hred16(s2[r]);
    mean[r] = su * (1.0f/128.0f);
    rstd[r] = rsqrtf(sq * (1.0f/128.0f) - mean[r]*mean[r] + 1e-5f);
  }
  #pragma unroll
  for (int nt = 0; nt < 8; ++nt) {
    int c = nt*16 + col;
    float g = vn_g[c], be = vn_b[c];
    #pragma unroll
    for (int r = 0; r < 8; ++r) {
      float y = (accU[nt][r] - mean[r]) * rstd[r] * g + be;
      accU[nt][r] = y;
      vt[(m0+r)*136 + c] = (_Float16)y;
    }
  }
  lds_fence();
  // ---- FFN: y + gelu(y@W1+b1)@W2+b2, fused via bounce ----
  v16h Afr[4];
  #pragma unroll
  for (int kt = 0; kt < 4; ++kt) Afr[kt] = load_af(vt + (size_t)brow*136 + kt*32);
  v8f accF[8] = {};
  #pragma unroll
  for (int p = 0; p < 8; ++p) {
    v8f aI[2] = {};
    #pragma unroll
    for (int kt = 0; kt < 4; ++kt) {
      v16h Bp[2];
      #pragma unroll
      for (int e = 0; e < 2; ++e) Bp[e] = load_bf(vf1Wt, 128, 2*p+e, kt*32);
      #pragma unroll
      for (int e = 0; e < 2; ++e) aI[e] = wmma(Afr[kt], Bp[e], aI[e]);
    }
    #pragma unroll
    for (int e = 0; e < 2; ++e) {
      float fb = vf1_b[(2*p+e)*16 + col];
      #pragma unroll
      for (int r = 0; r < 8; ++r)
        bc[(m0+r)*40 + e*16 + col] = (_Float16)gelu_f(aI[e][r] + fb);
    }
    lds_fence();
    v16h A2 = load_af(bc + (size_t)brow*40);
    v16h Bf[8];
    #pragma unroll
    for (int nt = 0; nt < 8; ++nt) Bf[nt] = load_bf(vf2Wt, 256, nt, p*32);
    #pragma unroll
    for (int nt = 0; nt < 8; ++nt) accF[nt] = wmma(A2, Bf[nt], accF[nt]);
  }
  #pragma unroll
  for (int nt = 0; nt < 8; ++nt) {
    int c = nt*16 + col;
    float fb = vf2_b[c];
    #pragma unroll
    for (int r = 0; r < 8; ++r)
      vout[((size_t)(bb*16+m0+r)*Nn + node)*Hd + c] = (_Float16)(accU[nt][r] + accF[nt][r] + fb);
  }
}

// ---------------------------------------------------------------------------
// hex: deflate(masked mean gather v x6)@W+b -> LN -> FFN (fused)
// ---------------------------------------------------------------------------
__global__ __launch_bounds__(256) void k_hex_upd(
    const _Float16* __restrict__ v16, _Float16* __restrict__ h16,
    const _Float16* __restrict__ defWt, const float* __restrict__ def_b,
    const float* __restrict__ hn_g, const float* __restrict__ hn_b,
    const _Float16* __restrict__ hf1Wt, const float* __restrict__ hf1_b,
    const _Float16* __restrict__ hf2Wt, const float* __restrict__ hf2_b,
    const int* __restrict__ h2v) {
  __shared__ _Float16 vtile[8][16*136];
  __shared__ _Float16 bnc[8][16*40];
  int wave = threadIdx.x >> 5, lane = threadIdx.x & 31;
  int tile = blockIdx.x * 8 + wave;
  int bb = tile / Tt, hex = tile - bb * Tt;
  int brow = lane & 15, m0 = (lane & 16) ? 8 : 0, col = lane & 15;
  _Float16* vt = &vtile[wave][0];
  _Float16* bc = &bnc[wave][0];
  prefetch_w(defWt, 128);

  int vi[6]; int cnt = 0;
  #pragma unroll
  for (int j = 0; j < 6; ++j) {
    vi[j] = __builtin_amdgcn_readfirstlane(h2v[hex*6 + j]);
    cnt += (vi[j] >= 0);
  }
  float rc = 1.0f / (float)(cnt > 1 ? cnt : 1);

  v8f accD[8] = {};
  #pragma unroll
  for (int kt = 0; kt < 4; ++kt) {
    v16h A = {};
    #pragma unroll
    for (int j = 0; j < 6; ++j)
      if (vi[j] >= 0)
        A = A + load_af(v16 + ((size_t)(bb*16+brow)*Nn + vi[j])*Hd + kt*32);
    v16h Bf[8];
    #pragma unroll
    for (int nt = 0; nt < 8; ++nt) Bf[nt] = load_bf(defWt, 128, nt, kt*32);
    #pragma unroll
    for (int nt = 0; nt < 8; ++nt) accD[nt] = wmma(A, Bf[nt], accD[nt]);
  }
  #pragma unroll
  for (int nt = 0; nt < 8; ++nt) {
    int c = nt*16 + col;
    float db = def_b[c];
    #pragma unroll
    for (int r = 0; r < 8; ++r)
      accD[nt][r] = accD[nt][r]*rc + db
                  + (float)h16[((size_t)(bb*16+m0+r)*Tt + hex)*Hd + c];
  }
  float s1[8] = {}, s2[8] = {};
  #pragma unroll
  for (int nt = 0; nt < 8; ++nt)
    #pragma unroll
    for (int r = 0; r < 8; ++r) { float v = accD[nt][r]; s1[r] += v; s2[r] += v*v; }
  float mean[8], rstd[8];
  #pragma unroll
  for (int r = 0; r < 8; ++r) {
    float su = hred16(s1[r]), sq = hred16(s2[r]);
    mean[r] = su * (1.0f/128.0f);
    rstd[r] = rsqrtf(sq * (1.0f/128.0f) - mean[r]*mean[r] + 1e-5f);
  }
  #pragma unroll
  for (int nt = 0; nt < 8; ++nt) {
    int c = nt*16 + col;
    float g = hn_g[c], be = hn_b[c];
    #pragma unroll
    for (int r = 0; r < 8; ++r) {
      float y = (accD[nt][r] - mean[r]) * rstd[r] * g + be;
      accD[nt][r] = y;
      vt[(m0+r)*136 + c] = (_Float16)y;
    }
  }
  lds_fence();
  v16h Afr[4];
  #pragma unroll
  for (int kt = 0; kt < 4; ++kt) Afr[kt] = load_af(vt + (size_t)brow*136 + kt*32);
  v8f accF[8] = {};
  #pragma unroll
  for (int p = 0; p < 8; ++p) {
    v8f aI[2] = {};
    #pragma unroll
    for (int kt = 0; kt < 4; ++kt) {
      v16h Bp[2];
      #pragma unroll
      for (int e = 0; e < 2; ++e) Bp[e] = load_bf(hf1Wt, 128, 2*p+e, kt*32);
      #pragma unroll
      for (int e = 0; e < 2; ++e) aI[e] = wmma(Afr[kt], Bp[e], aI[e]);
    }
    #pragma unroll
    for (int e = 0; e < 2; ++e) {
      float fb = hf1_b[(2*p+e)*16 + col];
      #pragma unroll
      for (int r = 0; r < 8; ++r)
        bc[(m0+r)*40 + e*16 + col] = (_Float16)gelu_f(aI[e][r] + fb);
    }
    lds_fence();
    v16h A2 = load_af(bc + (size_t)brow*40);
    v16h Bf[8];
    #pragma unroll
    for (int nt = 0; nt < 8; ++nt) Bf[nt] = load_bf(hf2Wt, 256, nt, p*32);
    #pragma unroll
    for (int nt = 0; nt < 8; ++nt) accF[nt] = wmma(A2, Bf[nt], accF[nt]);
  }
  #pragma unroll
  for (int nt = 0; nt < 8; ++nt) {
    int c = nt*16 + col;
    float fb = hf2_b[c];
    #pragma unroll
    for (int r = 0; r < 8; ++r)
      h16[((size_t)(bb*16+m0+r)*Tt + hex)*Hd + c] = (_Float16)(accD[nt][r] + accF[nt][r] + fb);
  }
}

// ---------------------------------------------------------------------------
// heads (tiny FLOPs -> VALU) + pooling
// ---------------------------------------------------------------------------
__global__ __launch_bounds__(64) void k_edge(
    const float* __restrict__ ef, const float* __restrict__ pW, const float* __restrict__ pB,
    const float* __restrict__ mW, const float* __restrict__ mB, _Float16* __restrict__ cmb) {
  __shared__ float ep[72][64];
  int b = blockIdx.x, n = threadIdx.x;
  for (int e = 0; e < 72; ++e) {
    const float* f = ef + ((size_t)b*72 + e)*8;
    float a = pB[n];
    #pragma unroll
    for (int k = 0; k < 8; ++k) a += f[k]*pW[k*64 + n];
    ep[e][n] = a;
  }
  __syncthreads();
  float s = 0.0f;
  for (int e = 0; e < 72; ++e) {
    float a = mB[n];
    for (int k = 0; k < 64; ++k) a += ep[e][k]*mW[k*64 + n];
    s += gelu_f(a);
  }
  cmb[(size_t)b*448 + 256 + n] = (_Float16)(s * (1.0f/72.0f));
}

__global__ __launch_bounds__(128) void k_player(
    const float* __restrict__ pf, const float* __restrict__ W1, const float* __restrict__ B1,
    const float* __restrict__ W2, const float* __restrict__ B2, _Float16* __restrict__ cmb) {
  __shared__ float t1[128];
  int b = blockIdx.x, n = threadIdx.x;
  const float* f = pf + (size_t)b*64;
  float a = B1[n];
  #pragma unroll
  for (int k = 0; k < 64; ++k) a += f[k]*W1[k*128 + n];
  t1[n] = gelu_f(a);
  __syncthreads();
  float a2 = B2[n];
  for (int k = 0; k < 128; ++k) a2 += t1[k]*W2[k*128 + n];
  cmb[(size_t)b*448 + 320 + n] = (_Float16)gelu_f(a2);
}

__global__ __launch_bounds__(256) void k_pool(
    const _Float16* __restrict__ h16, const _Float16* __restrict__ v16,
    _Float16* __restrict__ cmb) {
  int idx = blockIdx.x*256 + threadIdx.x;       // b*256 + c2
  if (idx >= Bsz*256) return;
  int b = idx >> 8, c2 = idx & 255;
  float s = 0.0f;
  if (c2 < 128) {
    for (int t = 0; t < Tt; ++t) s += (float)h16[((size_t)b*Tt + t)*Hd + c2];
    cmb[(size_t)b*448 + c2] = (_Float16)(s * (1.0f/19.0f));
  } else {
    int c = c2 - 128;
    for (int t = 0; t < Nn; ++t) s += (float)v16[((size_t)b*Nn + t)*Hd + c];
    cmb[(size_t)b*448 + 128 + c] = (_Float16)(s * (1.0f/54.0f));
  }
}

// ---------------------------------------------------------------------------
// output MLP: gelu(cmb@W1+b1)@W2+b2 -> f32 out, fused WMMA
// ---------------------------------------------------------------------------
__global__ __launch_bounds__(256) void k_out(
    const _Float16* __restrict__ cmb,
    const _Float16* __restrict__ w1t, const float* __restrict__ b1,
    const _Float16* __restrict__ w2t, const float* __restrict__ b2,
    float* __restrict__ out) {
  __shared__ _Float16 bnc[8][16*40];
  int wave = threadIdx.x >> 5, lane = threadIdx.x & 31;
  int mt = blockIdx.x * 8 + wave;               // 0..127 M-tiles over batch
  int brow = lane & 15, m0 = (lane & 16) ? 8 : 0, col = lane & 15;
  _Float16* bc = &bnc[wave][0];
  prefetch_w(w1t, 448);
  prefetch_w(w2t, 256);
  const _Float16* arow = cmb + (size_t)(mt*16 + brow)*448;
  v8f accO[16] = {};
  #pragma unroll
  for (int p = 0; p < 8; ++p) {                 // 16 inter n-tiles in pairs
    v8f aI[2] = {};
    #pragma unroll
    for (int kt = 0; kt < 14; ++kt) {
      v16h A = load_af(arow + kt*32);
      v16h Bp[2];
      #pragma unroll
      for (int e = 0; e < 2; ++e) Bp[e] = load_bf(w1t, 448, 2*p+e, kt*32);
      #pragma unroll
      for (int e = 0; e < 2; ++e) aI[e] = wmma(A, Bp[e], aI[e]);
    }
    #pragma unroll
    for (int e = 0; e < 2; ++e) {
      float fb = b1[(2*p+e)*16 + col];
      #pragma unroll
      for (int r = 0; r < 8; ++r)
        bc[(m0+r)*40 + e*16 + col] = (_Float16)gelu_f(aI[e][r] + fb);
    }
    lds_fence();
    v16h A2 = load_af(bc + (size_t)brow*40);
    #pragma unroll
    for (int g = 0; g < 2; ++g) {               // nt 0..15 in two groups of 8
      v16h Bf[8];
      #pragma unroll
      for (int i = 0; i < 8; ++i) Bf[i] = load_bf(w2t, 256, g*8 + i, p*32);
      #pragma unroll
      for (int i = 0; i < 8; ++i) accO[g*8 + i] = wmma(A2, Bf[i], accO[g*8 + i]);
    }
  }
  #pragma unroll
  for (int nt = 0; nt < 16; ++nt) {
    int c = nt*16 + col;
    float bb2 = b2[c];
    #pragma unroll
    for (int r = 0; r < 8; ++r)
      out[(size_t)(mt*16 + m0 + r)*256 + c] = accO[nt][r] + bb2;
  }
}

// ---------------------------------------------------------------------------
extern "C" void kernel_launch(void* const* d_in, const int* in_sizes, int n_in,
                              void* d_out, int out_size, void* d_ws, size_t ws_size,
                              hipStream_t stream) {
  (void)in_sizes; (void)n_in; (void)out_size; (void)ws_size;
  _Float16* ws = (_Float16*)d_ws;
  auto F = [&](int i) { return (const float*)d_in[i]; };
  auto I = [&](int i) { return (const int*)d_in[i]; };

  // 1) convert weight matrices -> f16 transposed in workspace
  ConvArgs ca; int ci = 0;
  for (int L = 0; L < 4; ++L) {
    size_t base = OFF_WBLK + (size_t)L * WBLK_SZ;
    int bi = IDX_BLK0 + 20*L;
    ca.d[ci++] = { F(bi+INF_W), (unsigned)(base+WO_INF), 384, 128 };
    ca.d[ci++] = { F(bi+MSG_W), (unsigned)(base+WO_MSG), 128, 128 };
    ca.d[ci++] = { F(bi+UPD_W), (unsigned)(base+WO_UPD), 256, 128 };
    ca.d[ci++] = { F(bi+DEF_W), (unsigned)(base+WO_DEF), 128, 128 };
    ca.d[ci++] = { F(bi+VF1_W), (unsigned)(base+WO_VF1), 128, 256 };
    ca.d[ci++] = { F(bi+VF2_W), (unsigned)(base+WO_VF2), 256, 128 };
    ca.d[ci++] = { F(bi+HF1_W), (unsigned)(base+WO_HF1), 128, 256 };
    ca.d[ci++] = { F(bi+HF2_W), (unsigned)(base+WO_HF2), 256, 128 };
  }
  ca.d[ci++] = { F(IDX_OUT1_W), (unsigned)OFF_OUT1, 448, 256 };
  ca.d[ci++] = { F(IDX_OUT2_W), (unsigned)OFF_OUT2, 256, 256 };
  k_convert<<<NCONV, 256, 0, stream>>>(ca, ws);

  // 2) input projections
  size_t ptot = (size_t)Bsz*Tt*Hd + (size_t)Bsz*Nn*Hd;
  k_proj<<<(int)((ptot + 255)/256), 256, 0, stream>>>(
      F(IDX_HEXF), F(IDX_VERTF), F(IDX_HPROJ_W), F(IDX_HPROJ_B),
      F(IDX_VPROJ_W), F(IDX_VPROJ_B), ws + OFF_H16, ws + OFF_V16);

  // 3) message-passing blocks
  for (int L = 0; L < 4; ++L) {
    size_t base = OFF_WBLK + (size_t)L * WBLK_SZ;
    int bi = IDX_BLK0 + 20*L;
    k_inflate<<<(Bsz/16)*Nn/8, 256, 0, stream>>>(
        ws + OFF_H16, ws + OFF_V16, ws + OFF_V1,
        ws + base + WO_INF, F(bi+INF_B), I(IDX_V2H));
    k_vertex_upd<<<(Bsz/16)*Nn/8, 256, 0, stream>>>(
        ws + OFF_V1, ws + OFF_V16,
        ws + base + WO_MSG, F(bi+MSG_B),
        ws + base + WO_UPD, F(bi+UPD_B),
        F(bi+VN_G), F(bi+VN_B),
        ws + base + WO_VF1, F(bi+VF1_B),
        ws + base + WO_VF2, F(bi+VF2_B), I(IDX_ADJ));
    k_hex_upd<<<(Bsz/16)*Tt/8, 256, 0, stream>>>(
        ws + OFF_V16, ws + OFF_H16,
        ws + base + WO_DEF, F(bi+DEF_B),
        F(bi+HN_G), F(bi+HN_B),
        ws + base + WO_HF1, F(bi+HF1_B),
        ws + base + WO_HF2, F(bi+HF2_B), I(IDX_H2V));
  }

  // 4) heads + pooling -> combined (B,448) f16
  k_edge<<<Bsz, 64, 0, stream>>>(F(IDX_EDGEF), F(IDX_EPROJ_W), F(IDX_EPROJ_B),
                                 F(IDX_EMLP_W), F(IDX_EMLP_B), ws + OFF_CMB);
  k_player<<<Bsz, 128, 0, stream>>>(F(IDX_PLAYERF), F(IDX_P1_W), F(IDX_P1_B),
                                    F(IDX_P2_W), F(IDX_P2_B), ws + OFF_CMB);
  k_pool<<<Bsz, 256, 0, stream>>>(ws + OFF_H16, ws + OFF_V16, ws + OFF_CMB);

  // 5) output MLP (WMMA, fused gelu)
  k_out<<<(Bsz/16)/8, 256, 0, stream>>>(ws + OFF_CMB, ws + OFF_OUT1, F(IDX_OUT1_B),
                                        ws + OFF_OUT2, F(IDX_OUT2_B), (float*)d_out);
}
